// BiLevelRoutingAttention_26688926777430
// MI455X (gfx1250) — matrix operand order repaired and checked by reading, not compiled, not run on hardware
//
#include <hip/hip_runtime.h>

typedef __bf16 bf16_t;
typedef bf16_t v16bf __attribute__((ext_vector_type(16)));
typedef float  v8f   __attribute__((ext_vector_type(8)));
typedef unsigned int u32x4 __attribute__((ext_vector_type(4)));
typedef int i32x8 __attribute__((ext_vector_type(8)));
typedef int i32x4 __attribute__((ext_vector_type(4)));

#define WMMA_BF16(a, b, c) \
  __builtin_amdgcn_wmma_f32_16x16x32_bf16(false, (a), false, (b), (short)0, (c), false, false)

constexpr int kB    = 2;
constexpr int kC    = 256;
constexpr int kNh   = 8;
constexpr int kHd   = 32;
constexpr int kSsp  = 3136;   // 56*56 spatial tokens
constexpr int kSpad = 3200;   // sequence padded: 25 groups x 8 qtiles x 16
constexpr int kAor  = 3152;   // attention-out rows per batch (mult of 16)
constexpr int kNblk = 50;
constexpr float kScale = 0.0625f;  // 256^-0.5

// ---- helpers -------------------------------------------------------------

// A-operand 16-bit layout: per lane two contiguous 8-element segments at
// +d0 and +d0+16 (d0 = 0 for lanes 0-15, 8 for lanes 16-31).
__device__ inline v16bf ld_bf16_splitA(const bf16_t* p, int d0) {
  union { uint4 u[2]; v16bf v; } U;
  U.u[0] = *(const uint4*)(p + d0);
  U.u[1] = *(const uint4*)(p + d0 + 16);
  return U.v;
}
// A-operand (V^T): 16 contiguous bf16 per lane, k = 16h + s.
__device__ inline v16bf ld_bf16_16(const bf16_t* p) {
  union { uint4 u[2]; v16bf v; } U;
  U.u[0] = *(const uint4*)(p);
  U.u[1] = *(const uint4*)(p + 8);
  return U.v;
}
__device__ inline void cvt8(v16bf& a, int off, float4 u, float4 v) {
  a[off + 0] = (bf16_t)u.x; a[off + 1] = (bf16_t)u.y;
  a[off + 2] = (bf16_t)u.z; a[off + 3] = (bf16_t)u.w;
  a[off + 4] = (bf16_t)v.x; a[off + 5] = (bf16_t)v.y;
  a[off + 6] = (bf16_t)v.z; a[off + 7] = (bf16_t)v.w;
}

// TDM 2D tile load: `rows` rows of `row_units` 8-byte units, global row
// stride `stride_units` (8-byte units), packed contiguously into LDS at
// byte offset lds_off.  D# per CDNA5 ISA sec 8.3/8.4 (group0 + group1,
// groups 2/3 zero since tile_dim2 = 0).  Toolchain uses the 6-arg builtin:
// (g0, g1, g2, g3, extra, cpol) -- extra passed zero-filled.
__device__ inline void tdm_load_tile(unsigned lds_off, const void* gaddr,
                                     unsigned row_units, unsigned rows,
                                     unsigned stride_units) {
  const unsigned long long ga = (unsigned long long)(uintptr_t)gaddr;
  u32x4 g0 = { 1u,                                    // count=1 valid descriptor
               lds_off,                               // lds_addr (bytes)
               (unsigned)ga,                          // global_addr[31:0]
               (unsigned)((ga >> 32) & 0x01FFFFFFu) | (2u << 30) }; // [56:32] | type=2
  i32x8 g1 = { (int)(3u << 16),                                   // data_size = 8B
               (int)((row_units & 0xFFFFu) << 16),                // tensor_dim0 lo16
               (int)(((row_units >> 16) & 0xFFFFu) |
                     ((rows & 0xFFFFu) << 16)),                   // td0 hi | tensor_dim1 lo
               (int)(((rows >> 16) & 0xFFFFu) |
                     ((row_units & 0xFFFFu) << 16)),              // td1 hi | tile_dim0
               (int)(rows & 0xFFFFu),                             // tile_dim1 (tile_dim2=0)
               (int)stride_units,                                 // tensor_dim0_stride lo32
               0, 0 };
  i32x4 z4 = { 0, 0, 0, 0 };
  i32x8 z8 = { 0, 0, 0, 0, 0, 0, 0, 0 };
  __builtin_amdgcn_tensor_load_to_lds(g0, g1, z4, z4, z8, 0);
}

// ---- kernel 1: QKV projection + window permute + bf16 pack ---------------
// GEMM: [B*kSpad, 256] x Wp^T[256, 768]. One wave = one 16x16 output tile.
// waves = (2*3200/16) * (768/16) = 400*48 = 19200 -> 2400 blocks of 256.
__global__ void __launch_bounds__(256) qkv_win_kernel(
    const float* __restrict__ x, const float* __restrict__ cls,
    const float* __restrict__ Wp, const float* __restrict__ bp,
    bf16_t* __restrict__ qb, bf16_t* __restrict__ kb,
    bf16_t* __restrict__ vbt, float* __restrict__ vsp)
{
  const int wave = (blockIdx.x * blockDim.x + threadIdx.x) >> 5;
  const int lane = threadIdx.x & 31;
  const int half = lane >> 4, lm = lane & 15;
  const int ntile = wave % 48;
  const int mtile = wave / 48;
  if (mtile >= (kB * kSpad) / 16) return;  // never taken (exact grid)
  const int nG = ntile * 16 + lm;          // output column 0..767

  const int growA = mtile * 16 + lm;
  const int bA = growA / kSpad;
  const int tA = growA % kSpad;
  const float* rowp = nullptr;
  if (tA < kSsp)       rowp = x + ((size_t)bA * kSsp + tA) * kC;
  else if (tA == kSsp) rowp = cls + (size_t)bA * kC;
  const float* wrow = Wp + (size_t)nG * kC;

  v8f acc = {};
  for (int kc = 0; kc < 8; ++kc) {
    const int s0 = kc * 32 + half * 8;
    v16bf a;
    if (rowp) {
      cvt8(a, 0, *(const float4*)(rowp + s0),      *(const float4*)(rowp + s0 + 4));
      cvt8(a, 8, *(const float4*)(rowp + s0 + 16), *(const float4*)(rowp + s0 + 20));
    } else {
      for (int e = 0; e < 16; ++e) a[e] = (bf16_t)0.0f;
    }
    const int kb0 = kc * 32 + half * 8;
    __builtin_prefetch(wrow + kb0 + 32, 0, 1);
    v16bf bm;
    cvt8(bm, 0, *(const float4*)(wrow + kb0),     *(const float4*)(wrow + kb0 + 4));
    cvt8(bm, 8, *(const float4*)(wrow + kb0 + 16), *(const float4*)(wrow + kb0 + 20));
    acc = WMMA_BF16(a, bm, acc);
  }
  const float bias = bp[nG];
  const int part = nG >> 8, c = nG & 255, head = c >> 5, d = c & 31;
  for (int r = 0; r < 8; ++r) {
    const int m = r + half * 8;
    const int grow = mtile * 16 + m;
    const int b = grow / kSpad, t = grow % kSpad;
    float val = acc[r] + bias;
    int s;
    if (t < kSsp) {                              // window permute
      const int h = t / 56, w = t - h * 56;
      s = ((h >> 3) * 7 + (w >> 3)) * 64 + (h & 7) * 8 + (w & 7);
    } else if (t == kSsp) {
      s = kSsp;                                  // cls token
    } else {
      s = t; val = 0.0f;                         // padding row: write zeros
    }
    const bf16_t bv = (bf16_t)val;
    const size_t bh = (size_t)(b * kNh + head);
    if (part == 0)      qb[(bh * kSpad + s) * kHd + d] = bv;
    else if (part == 1) kb[(bh * kSpad + s) * kHd + d] = bv;
    else {
      vbt[(bh * kHd + d) * kSpad + s] = bv;      // V stored transposed [d][s]
      if (t < kSsp) vsp[((size_t)b * kSsp + t) * kC + c] = val;
    }
  }
}

// ---- kernel 2: LePE depthwise 3x3 conv (SAME, NHWC) ----------------------
__global__ void __launch_bounds__(256) lepe_kernel(
    const float* __restrict__ vsp, const float* __restrict__ lw,
    const float* __restrict__ lb, float* __restrict__ lep)
{
  const size_t idx = (size_t)blockIdx.x * blockDim.x + threadIdx.x;
  const int c = (int)(idx & 255);
  const size_t r = idx >> 8;
  const int w = (int)(r % 56);
  const size_t r2 = r / 56;
  const int h = (int)(r2 % 56);
  const int b = (int)(r2 / 56);
  if (b >= kB) return;
  float acc = lb[c];
  for (int kh = 0; kh < 3; ++kh) {
    const int ih = h + kh - 1;
    if (ih < 0 || ih >= 56) continue;
    for (int kw = 0; kw < 3; ++kw) {
      const int iw = w + kw - 1;
      if (iw < 0 || iw >= 56) continue;
      acc += vsp[(((size_t)b * kSsp) + ih * 56 + iw) * kC + c] *
             lw[(kh * 3 + kw) * kC + c];
    }
  }
  lep[idx] = acc;
}

// ---- kernel 3: flash attention, TDM-staged K/V in LDS --------------------
// One block = 8 waves sharing one (b,head); each wave owns one 16-query tile
// (block covers 8 consecutive qtiles = 128 queries).  K/V 32-key tiles are
// DMA'd into LDS once per block by the Tensor Data Mover and consumed by all
// 8 waves -> 8x less L2 traffic than per-wave loads.
// grid = 16 (b,head) * 25 qgroups = 400 blocks of 256; dynLDS = 4 KB.
__global__ void __launch_bounds__(256) attn_kernel(
    const bf16_t* __restrict__ qb, const bf16_t* __restrict__ kb,
    const bf16_t* __restrict__ vbt, const int* __restrict__ mask,
    float* __restrict__ ao)
{
  extern __shared__ bf16_t smem[];           // [0,2048)B: K tile  [key][d]
  bf16_t* smemV = smem + 1024;               // [2048,4096)B: V^T tile [d][key]

  const int lane = threadIdx.x & 31;
  const int wib  = threadIdx.x >> 5;         // wave in block: 0..7
  const int half = lane >> 4, lm = lane & 15;
  const int bh = blockIdx.x / 25;            // 0..15
  const int qg = blockIdx.x % 25;            // query group
  const int b = bh >> 3, head = bh & 7;
  const int qtile = qg * 8 + wib;            // 0..199

  // Q^T as B-operand: lane = query, elems = head-dims in split 8+8 layout (d0, d0+16).
  const int qs = qtile * 16 + lm;
  const v16bf Bq = ld_bf16_splitA(qb + ((size_t)bh * kSpad + qs) * kHd, half * 8);

  v8f olo = {}, ohi = {};
  float mi = -3.0e38f, li = 0.0f;
  const int blkq = ((qtile >> 2) < 49) ? (qtile >> 2) : 49;
  const int rA = (2 * qg < 49) ? 2 * qg : 49;        // block's two query blocks
  const int rB = (2 * qg + 1 < 49) ? 2 * qg + 1 : 49;
  const int* mrow  = mask + (b * kNblk + blkq) * kNblk;
  const int* mrowA = mask + (b * kNblk + rA) * kNblk;
  const int* mrowB = mask + (b * kNblk + rB) * kNblk;

  const bf16_t* kbase = kb + (size_t)bh * kSpad * kHd;
  const bf16_t* vbase = vbt + (size_t)bh * kHd * kSpad;

  for (int kt = 0; kt < 99; ++kt) {          // key tiles of 32; kt=99 is all pad
    const int blkk = (kt < 98) ? (kt >> 1) : 49;
    if ((mrowA[blkk] | mrowB[blkk]) == 0) continue;  // block-uniform skip
    const int ks = kt * 32;

    if (wib == 0) {                          // wave 0 drives the TDM
      // K tile: 32 rows x 64B, contiguous (row stride = 8 units of 8B)
      tdm_load_tile(0,    kbase + (size_t)ks * kHd, 8, 32, 8);
      // V^T tile: 32 rows x 64B at row stride kSpad*2B = 800 units
      tdm_load_tile(2048, vbase + ks,               8, 32, kSpad * 2 / 8);
      __builtin_amdgcn_s_wait_tensorcnt(0);
    }
    __syncthreads();                         // publish LDS tiles to all waves

    if (mrow[blkk] != 0) {                   // wave-uniform: EXEC stays all-1s
      const v16bf A0  = ld_bf16_splitA(smem  + lm * 32,        half * 8);
      const v16bf A1  = ld_bf16_splitA(smem  + (16 + lm) * 32, half * 8);
      const v16bf Vlo = ld_bf16_16(smemV + lm * 32 + half * 16);
      const v16bf Vhi = ld_bf16_16(smemV + (16 + lm) * 32 + half * 16);
      v8f z = {};
      v8f s0 = WMMA_BF16(A0, Bq, z);         // [keys 0..15  x queries]
      v8f s1 = WMMA_BF16(A1, Bq, z);         // [keys 16..31 x queries]

      if (kt == 98) {                        // cls (s=3136) + padding keys
        for (int r = 0; r < 8; ++r) {
          s0[r] = (r == 0 && half == 0) ? s0[r] * kScale : -1.0e30f;
          s1[r] = -1.0e30f;
        }
      } else {
        for (int r = 0; r < 8; ++r) { s0[r] *= kScale; s1[r] *= kScale; }
      }

      float mx = -3.0e38f;
      for (int r = 0; r < 8; ++r) { mx = fmaxf(mx, s0[r]); mx = fmaxf(mx, s1[r]); }
      mx = fmaxf(mx, __shfl_xor(mx, 16, 32));
      const float nm = fmaxf(mi, mx);
      const float corr = __expf(mi - nm);
      float rs = 0.0f;
      for (int r = 0; r < 8; ++r) {
        s0[r] = __expf(s0[r] - nm);
        s1[r] = __expf(s1[r] - nm);
        rs += s0[r] + s1[r];
      }
      rs += __shfl_xor(rs, 16, 32);
      li = li * corr + rs;
      mi = nm;
      for (int r = 0; r < 8; ++r) { olo[r] *= corr; ohi[r] *= corr; }

      // P^T rebuilt into B-operand layout with 16 half-swap shuffles.
      v16bf Pt;
      for (int r = 0; r < 8; ++r) {
        const float x0 = __shfl_xor(s0[r], 16, 32);
        const float x1 = __shfl_xor(s1[r], 16, 32);
        Pt[r]     = (bf16_t)(half ? x1 : s0[r]);
        Pt[r + 8] = (bf16_t)(half ? s1[r] : x0);
      }
      olo = WMMA_BF16(Vlo, Pt, olo);
      ohi = WMMA_BF16(Vhi, Pt, ohi);
    }
    __syncthreads();                         // LDS reuse next iteration
  }

  const float rl = 1.0f / li;
  const int sq = qtile * 16 + lm;
  if (sq <= kSsp) {
    int t;
    if (sq < kSsp) {                         // inverse window permute
      const int blk = sq >> 6, off = sq & 63;
      const int h = (blk / 7) * 8 + (off >> 3);
      const int w = (blk % 7) * 8 + (off & 7);
      t = h * 56 + w;
    } else {
      t = kSsp;
    }
    float* orow = ao + ((size_t)b * kAor + t) * kC + head * kHd;
    for (int r = 0; r < 8; ++r) {
      orow[r + half * 8]      = olo[r] * rl;
      orow[r + half * 8 + 16] = ohi[r] * rl;
    }
  }
}

// ---- kernel 4: output projection (attn_out + lepe) @ Wo^T + bo -----------
// waves = 2 * (3152/16) * (256/16) = 6304 -> 788 blocks.
__global__ void __launch_bounds__(256) proj_kernel(
    const float* __restrict__ ao, const float* __restrict__ lepe,
    const float* __restrict__ Wo, const float* __restrict__ bo,
    float* __restrict__ out)
{
  const int wave = (blockIdx.x * blockDim.x + threadIdx.x) >> 5;
  const int lane = threadIdx.x & 31;
  const int half = lane >> 4, lm = lane & 15;
  const int ntile = wave & 15;
  const int mt = wave >> 4;
  const int b = mt / 197, mtile = mt % 197;
  if (b >= kB) return;                       // never taken (exact grid)
  const int nG = ntile * 16 + lm;
  const int t = mtile * 16 + lm;
  const float* arow = ao + ((size_t)b * kAor + t) * kC;
  const float* lrow = (t < kSsp) ? (lepe + ((size_t)b * kSsp + t) * kC) : nullptr;
  const float* wrow = Wo + (size_t)nG * kC;

  v8f acc = {};
  for (int kc = 0; kc < 8; ++kc) {
    const int s0 = kc * 32 + half * 8;
    float4 f0 = *(const float4*)(arow + s0);
    float4 f1 = *(const float4*)(arow + s0 + 4);
    float4 f2 = *(const float4*)(arow + s0 + 16);
    float4 f3 = *(const float4*)(arow + s0 + 20);
    if (lrow) {
      const float4 g0 = *(const float4*)(lrow + s0);
      const float4 g1 = *(const float4*)(lrow + s0 + 4);
      const float4 g2 = *(const float4*)(lrow + s0 + 16);
      const float4 g3 = *(const float4*)(lrow + s0 + 20);
      f0.x += g0.x; f0.y += g0.y; f0.z += g0.z; f0.w += g0.w;
      f1.x += g1.x; f1.y += g1.y; f1.z += g1.z; f1.w += g1.w;
      f2.x += g2.x; f2.y += g2.y; f2.z += g2.z; f2.w += g2.w;
      f3.x += g3.x; f3.y += g3.y; f3.z += g3.z; f3.w += g3.w;
    }
    v16bf a;
    cvt8(a, 0, f0, f1);
    cvt8(a, 8, f2, f3);
    const int kb0 = kc * 32 + half * 8;
    v16bf bm;
    cvt8(bm, 0, *(const float4*)(wrow + kb0),     *(const float4*)(wrow + kb0 + 4));
    cvt8(bm, 8, *(const float4*)(wrow + kb0 + 16), *(const float4*)(wrow + kb0 + 20));
    acc = WMMA_BF16(a, bm, acc);
  }
  const float bias = bo[nG];
  for (int r = 0; r < 8; ++r) {
    const int tt = mtile * 16 + r + half * 8;
    const float v = acc[r] + bias;
    if (tt < kSsp)       out[((size_t)b * kSsp + tt) * kC + nG] = v;
    else if (tt == kSsp) out[(size_t)kB * kSsp * kC + (size_t)b * kC + nG] = v;
  }
}

// ---- launch --------------------------------------------------------------
extern "C" void kernel_launch(void* const* d_in, const int* in_sizes, int n_in,
                              void* d_out, int out_size, void* d_ws, size_t ws_size,
                              hipStream_t stream) {
  const float* x    = (const float*)d_in[0];
  const float* cls  = (const float*)d_in[1];
  const int*   mask = (const int*)d_in[2];
  const float* Wp   = (const float*)d_in[3];
  const float* bp   = (const float*)d_in[4];
  const float* lw   = (const float*)d_in[5];
  const float* lb   = (const float*)d_in[6];
  const float* Wo   = (const float*)d_in[7];
  const float* bo   = (const float*)d_in[8];
  float* out = (float*)d_out;

  (void)in_sizes; (void)n_in; (void)out_size; (void)ws_size;

  char* p = (char*)d_ws;
  auto take = [&p](size_t bytes) -> char* {
    char* r = p;
    p += (bytes + 255) & ~(size_t)255;
    return r;
  };
  const size_t nQKV = (size_t)kB * kNh * kSpad * kHd;          // 1,638,400 elems
  bf16_t* qb  = (bf16_t*)take(nQKV * sizeof(bf16_t));
  bf16_t* kbf = (bf16_t*)take(nQKV * sizeof(bf16_t));
  bf16_t* vbt = (bf16_t*)take(nQKV * sizeof(bf16_t));
  float*  vsp = (float*)take((size_t)kB * kSsp * kC * 4);
  float*  lep = (float*)take((size_t)kB * kSsp * kC * 4);
  float*  ao  = (float*)take((size_t)kB * kAor * kC * 4);

  qkv_win_kernel<<<2400, 256, 0, stream>>>(x, cls, Wp, bp, qb, kbf, vbt, vsp);
  lepe_kernel   <<<6272, 256, 0, stream>>>(vsp, lw, lb, lep);
  attn_kernel   <<<400,  256, 4096, stream>>>(qb, kbf, vbt, mask, ao);
  proj_kernel   <<<788,  256, 0, stream>>>(ao, lep, Wo, bo, out);
}
